// LatentSpaceClustering_46797963657837
// MI455X (gfx1250) — matrix operand order repaired and checked by reading, not compiled, not run on hardware
//
#include <hip/hip_runtime.h>

typedef __attribute__((ext_vector_type(16))) _Float16 v16h;
typedef __attribute__((ext_vector_type(8)))  float    v8f;

#define N_PTS 1048576
#define H_DIM 64
#define K_CL  256
#define WAVES 8
#define RT    2                                  // row-tiles per wave
#define ROWS_PER_WAVE  (RT * 16)                 // 32
#define ROWS_PER_GROUP (WAVES * ROWS_PER_WAVE)   // 256
#define NGROUPS        (N_PTS / ROWS_PER_GROUP)  // 4096
#define GRID_BLOCKS    1024                      // 4 groups per block (grid-stride)

// LDS: Bhi 32KB + Blo 32KB = 64KB. c2 staged temporarily at offset 0 pre-fill.
#define LDS_BLO  32768
#define LDS_BYTES 65536

__global__ __launch_bounds__(256)
void kmeans_assign_wmma(const float* __restrict__ x,
                        const float* __restrict__ cen,
                        int* __restrict__ out)
{
    extern __shared__ char smem[];
    _Float16* Bhi = (_Float16*)smem;
    _Float16* Blo = (_Float16*)(smem + LDS_BLO);
    float*    c2s = (float*)smem;                // staging only, overwritten by Bhi

    const int tid  = threadIdx.x;
    const int wave = tid >> 5;
    const int lane = tid & 31;
    const int h    = lane >> 4;   // half-wave select
    const int nl   = lane & 15;   // row (A) / col (B,C) within 16x16 tile

    // ---- stage ||c||^2 (fp32 exact), then hoist each lane's 16 values to regs ----
    {
        float s = 0.f;
        const float* cp = cen + tid * H_DIM;
        #pragma unroll
        for (int k = 0; k < H_DIM; ++k) { float v = cp[k]; s += v * v; }
        c2s[tid] = s;
    }
    __syncthreads();
    float c2r[16];
    #pragma unroll
    for (int t = 0; t < 16; ++t) c2r[t] = c2s[t * 16 + nl];
    __syncthreads();

    // ---- once-per-block fill of all 256 centers as WMMA B operands (f16 hi+lo) ----
    // B 16-bit 32x16 layout: lane (h*16+col): e=0..7 -> K=8h+e ; e=8..15 -> K=16+8h+(e-8)
    for (int idx = tid; idx < K_CL * H_DIM; idx += 256) {
        int c = idx >> 6;                 // center 0..255
        int k = idx & 63;                 // dim
        float v = cen[idx];
        _Float16 vh = (_Float16)v;
        _Float16 vl = (_Float16)(v - (float)vh);
        int ks = k >> 5, kk = k & 31;
        int lb = (((kk >> 3) & 1) << 4) + (c & 15);
        int eb = (kk & 7) + ((kk >> 4) << 3);
        int pos = ((((c >> 4) << 1) + ks) * 32 + lb) * 16 + eb;
        Bhi[pos] = vh;
        Blo[pos] = vl;
    }
    __syncthreads();

    // ---- persistent loop over row-groups (no further block-wide syncs) ----
    for (long g = blockIdx.x; g < NGROUPS; g += gridDim.x) {
        const long rowBase = g * ROWS_PER_GROUP + (long)wave * ROWS_PER_WAVE;

        // prefetch next group's x rows while this group computes
        long gn = g + gridDim.x;
        if (gn < NGROUPS) {
            const float* nx = x + (gn * ROWS_PER_GROUP + tid) * H_DIM;
            __builtin_prefetch(nx, 0, 1);
            __builtin_prefetch(nx + 32, 0, 1);
        }

        // load + split A (fp32 -> f16 hi + f16 lo), 2 row-tiles x 2 K-steps
        // A 16-bit layout: lane (h*16+nl): row M = nl; elements:
        //   e = 0..7  -> K = ks*32 + 8h + e ;  e = 8..15 -> K = ks*32 + 16 + 8h + (e-8)
        v16h a_hi[RT][2], a_lo[RT][2];
        #pragma unroll
        for (int rt = 0; rt < RT; ++rt) {
            const float* xr = x + (rowBase + rt * 16 + nl) * H_DIM;
            #pragma unroll
            for (int ks = 0; ks < 2; ++ks) {
                const float* g1 = xr + ks * 32 + 8 * h;
                const float* g2 = g1 + 16;
                #pragma unroll
                for (int j = 0; j < 8; ++j) {
                    float v1 = g1[j];
                    _Float16 h1 = (_Float16)v1;
                    a_hi[rt][ks][j]     = h1;
                    a_lo[rt][ks][j]     = (_Float16)(v1 - (float)h1);
                    float v2 = g2[j];
                    _Float16 h2 = (_Float16)v2;
                    a_hi[rt][ks][8 + j] = h2;
                    a_lo[rt][ks][8 + j] = (_Float16)(v2 - (float)h2);
                }
            }
        }

        float best[2 * 8];
        int   bidx[2 * 8];
        #pragma unroll
        for (int v = 0; v < 16; ++v) { best[v] = 3.4e38f; bidx[v] = 0; }

        // partial unroll: 4 tiles in flight keeps VGPR pressure < 256
        // (full unroll caused s_set_vgpr_msb churn and occupancy loss)
        #pragma unroll 4
        for (int tile = 0; tile < 16; ++tile) {
            v8f acc0 = {}, acc1 = {};
            #pragma unroll
            for (int ks = 0; ks < 2; ++ks) {
                int pos = (((tile << 1) + ks) * 32 + lane) * 16;
                v16h bh = *(const v16h*)(Bhi + pos);
                v16h bl = *(const v16h*)(Blo + pos);
                acc0 = __builtin_amdgcn_wmma_f32_16x16x32_f16(false, a_hi[0][ks], false, bh,
                                                              (short)0, acc0, false, false);
                acc0 = __builtin_amdgcn_wmma_f32_16x16x32_f16(false, a_hi[0][ks], false, bl,
                                                              (short)0, acc0, false, false);
                acc0 = __builtin_amdgcn_wmma_f32_16x16x32_f16(false, a_lo[0][ks], false, bh,
                                                              (short)0, acc0, false, false);
                acc1 = __builtin_amdgcn_wmma_f32_16x16x32_f16(false, a_hi[1][ks], false, bh,
                                                              (short)0, acc1, false, false);
                acc1 = __builtin_amdgcn_wmma_f32_16x16x32_f16(false, a_hi[1][ks], false, bl,
                                                              (short)0, acc1, false, false);
                acc1 = __builtin_amdgcn_wmma_f32_16x16x32_f16(false, a_lo[1][ks], false, bh,
                                                              (short)0, acc1, false, false);
            }
            // score = ||c||^2 - 2 x.c  (monotone-equivalent to squared distance)
            int col = tile * 16 + nl;
            float cc = c2r[tile];
            #pragma unroll
            for (int v = 0; v < 8; ++v) {
                float s0 = cc - 2.0f * acc0[v];
                if (s0 < best[v])     { best[v]     = s0; bidx[v]     = col; }
                float s1 = cc - 2.0f * acc1[v];
                if (s1 < best[8 + v]) { best[8 + v] = s1; bidx[8 + v] = col; }
            }
        }

        // cross-lane argmin within each 16-lane half (columns spread mod 16)
        #pragma unroll
        for (int m = 1; m <= 8; m <<= 1) {
            #pragma unroll
            for (int v = 0; v < 16; ++v) {
                float ob = __shfl_xor(best[v], m, 32);
                int   oi = __shfl_xor(bidx[v], m, 32);
                if (ob < best[v] || (ob == best[v] && oi < bidx[v])) {
                    best[v] = ob; bidx[v] = oi;
                }
            }
        }

        // lane nl==0 of each half holds rows: rt0 -> rowBase + h*8 + v,
        //                                     rt1 -> rowBase + 16 + h*8 + v
        if (nl == 0) {
            long r0 = rowBase + h * 8;
            #pragma unroll
            for (int v = 0; v < 8; ++v) {
                out[r0 + v]      = bidx[v];
                out[r0 + 16 + v] = bidx[8 + v];
            }
        }
    }
}

extern "C" void kernel_launch(void* const* d_in, const int* in_sizes, int n_in,
                              void* d_out, int out_size, void* d_ws, size_t ws_size,
                              hipStream_t stream) {
    const float* x   = (const float*)d_in[0];   // (N, 64) fp32
    const float* cen = (const float*)d_in[1];   // (256, 64) fp32
    int* out = (int*)d_out;                     // (N, 1) int32 argmin indices

    dim3 grid(GRID_BLOCKS);
    dim3 block(256);                            // 8 wave32 waves
    hipLaunchKernelGGL(kmeans_assign_wmma, grid, block, LDS_BYTES, stream,
                       x, cen, out);
}
